// SelfAttentionModule_15204184228335
// MI455X (gfx1250) — compile-verified, matrix operand order checked by reading, not run on hardware
//
#include <hip/hip_runtime.h>
#include <hip/hip_bf16.h>

// ---------------------------------------------------------------------------
// Types / WMMA helpers (CDNA5 wave32, v_wmma_f32_16x16x32_bf16)
// ---------------------------------------------------------------------------
typedef __attribute__((ext_vector_type(16))) __bf16 v16bf;
typedef __attribute__((ext_vector_type(8)))  __bf16 v8bf;
typedef __attribute__((ext_vector_type(8)))  float  v8f;

union Frag {
    v16bf v;
    v8bf  h[2];
};

// Per-lane fragment load: elements {base[0..7], base[16..23]} (base already
// includes the half-wave K offset).  Both 16-byte halves must be 16B aligned.
__device__ __forceinline__ Frag load_frag(const __bf16* __restrict__ base) {
    Frag f;
    f.h[0] = *(const v8bf*)(base);
    f.h[1] = *(const v8bf*)(base + 16);
    return f;
}

__device__ __forceinline__ v8f wmma_bf16(const Frag& a, const Frag& b, v8f c) {
    return __builtin_amdgcn_wmma_f32_16x16x32_bf16(
        /*neg_a=*/false, a.v, /*neg_b=*/false, b.v,
        /*c_mod=*/(short)0, c, /*reuse_a=*/false, /*reuse_b=*/false);
}

// ---------------------------------------------------------------------------
// Problem constants
// ---------------------------------------------------------------------------
#define BATCH 8
#define CIN   512
#define HIN   96
#define WIN   96
#define HP    48
#define WP    48
#define NPIX  (HP * WP)           // 2304
#define BN_   (BATCH * NPIX)      // 18432
#define KD    256
#define VD    256
#define OD    512
#define HWOUT (HIN * WIN)         // 9216
#define BHW   (BATCH * HWOUT)     // 73728
#define EPS   1e-5f

// Workspace layout (bytes)
#define OFF_XP    ((size_t)0)                         // bf16 [BN_][CIN]
#define OFF_QK    (OFF_XP   + (size_t)BN_*CIN*2)      // bf16 [BN_][KD]
#define OFF_VALP  (OFF_QK   + (size_t)BN_*KD*2)       // bf16 [BN_][VD]
#define OFF_VALT  (OFF_VALP + (size_t)BN_*VD*2)       // bf16 [B][VD][NPIX]
#define OFF_CTX   (OFF_VALT + (size_t)BN_*VD*2)       // bf16 [BN_][VD]
#define OFF_CTXW  (OFF_CTX  + (size_t)BN_*VD*2)       // f32  [BN_][OD]
#define OFF_UP    (OFF_CTXW + (size_t)BN_*OD*4)       // bf16 [BHW][OD]
#define OFF_WKF   (OFF_UP   + (size_t)BHW*OD*2)       // bf16 [KD][CIN]
#define OFF_WVF   (OFF_WKF  + (size_t)KD*CIN*2)       // bf16 [VD][CIN]
#define OFF_WWF   (OFF_WVF  + (size_t)VD*CIN*2)       // bf16 [OD][VD]
#define OFF_WRF   (OFF_WWF  + (size_t)OD*VD*2)        // bf16 [OD][OD]
#define OFF_BK    (OFF_WRF  + (size_t)OD*OD*2)        // f32  [KD]
#define OFF_BR    (OFF_BK   + (size_t)KD*4)           // f32  [OD]

// ---------------------------------------------------------------------------
// 0) Fold BN into weights, cast all weights to bf16
// ---------------------------------------------------------------------------
__global__ void prep_weights(const float* __restrict__ Wk, const float* __restrict__ kg,
                             const float* __restrict__ kb, const float* __restrict__ km,
                             const float* __restrict__ kv, const float* __restrict__ Wv,
                             const float* __restrict__ Ww, const float* __restrict__ Wr,
                             const float* __restrict__ rg, const float* __restrict__ rb,
                             const float* __restrict__ rm, const float* __restrict__ rv,
                             __bf16* __restrict__ wkf, __bf16* __restrict__ wvf,
                             __bf16* __restrict__ wwf, __bf16* __restrict__ wrf,
                             float* __restrict__ bk, float* __restrict__ br) {
    for (size_t i = (size_t)blockIdx.x * blockDim.x + threadIdx.x;
         i < (size_t)OD * OD; i += (size_t)gridDim.x * blockDim.x) {
        if (i < KD) { float inv = kg[i] * rsqrtf(kv[i] + EPS); bk[i] = kb[i] - km[i] * inv; }
        if (i < OD) { float inv = rg[i] * rsqrtf(rv[i] + EPS); br[i] = rb[i] - rm[i] * inv; }
        if (i < (size_t)KD * CIN) {
            int k = (int)(i >> 9);                              // /512
            float invk = kg[k] * rsqrtf(kv[k] + EPS);
            wkf[i] = (__bf16)(Wk[i] * invk);
            wvf[i] = (__bf16)Wv[i];
            wwf[i] = (__bf16)Ww[i];                             // [OD][VD] = 131072 too
        }
        int o = (int)(i >> 9);
        float invr = rg[o] * rsqrtf(rv[o] + EPS);
        wrf[i] = (__bf16)(Wr[i] * invr);
    }
}

// ---------------------------------------------------------------------------
// 1) MaxPool 2x2 + NCHW -> (pixel, channel) bf16, via LDS transpose tile
//    grid(3, 48, 8*32), block 256 (16 x-positions x 16 channels)
// ---------------------------------------------------------------------------
__global__ void pool_kernel(const float* __restrict__ x, __bf16* __restrict__ xp) {
    __shared__ __bf16 t[16][17];
    int tx = threadIdx.x & 15, ty = threadIdx.x >> 4;
    int xt = blockIdx.x, yo = blockIdx.y;
    int b = blockIdx.z >> 5, ct = blockIdx.z & 31;
    int c = ct * 16 + ty;
    int xo = xt * 16 + tx;
    const float* p = x + (((size_t)b * CIN + c) * HIN + 2 * yo) * WIN + 2 * xo;
    float2 a0 = *(const float2*)p;
    float2 a1 = *(const float2*)(p + WIN);
    float m = fmaxf(fmaxf(a0.x, a0.y), fmaxf(a1.x, a1.y));
    t[ty][tx] = (__bf16)m;
    __syncthreads();
    int n = yo * WP + xt * 16 + ty;                 // pixel index in batch
    xp[((size_t)b * NPIX + n) * CIN + ct * 16 + tx] = t[tx][ty];
}

// ---------------------------------------------------------------------------
// 2) Fused qk / value 1x1 conv GEMM.  Wave = 16 pixels x 64 out-channels.
//    o < 256 -> ReLU(qk + folded-BN bias) -> QK[pix][o]
//    o >=256 -> value + bv               -> VALP[pix][o-256]
//    grid(BN_/16, 2), block 128 (4 waves)
// ---------------------------------------------------------------------------
__global__ void qkv_gemm(const __bf16* __restrict__ xp, const __bf16* __restrict__ wkf,
                         const __bf16* __restrict__ wvf, const float* __restrict__ bk,
                         const float* __restrict__ bv, __bf16* __restrict__ qk,
                         __bf16* __restrict__ valp) {
    int lane = threadIdx.x & 31, w = threadIdx.x >> 5;
    int p0 = blockIdx.x * 16;
    int obase = (blockIdx.y * 4 + w) * 64;
    int col = lane & 15;
    int h8 = (lane < 16) ? 0 : 8;
    int rbase = (lane >> 4) << 3;
    const __bf16* arow = xp + (size_t)(p0 + col) * CIN + h8;
    v8f acc[4] = {};
    for (int kt = 0; kt < CIN / 32; ++kt) {
        Frag a = load_frag(arow + kt * 32);
        for (int j = 0; j < 4; ++j) {
            int o = obase + j * 16 + col;
            const __bf16* brow = (o < KD) ? (wkf + (size_t)o * CIN)
                                          : (wvf + (size_t)(o - KD) * CIN);
            Frag bm = load_frag(brow + kt * 32 + h8);
            acc[j] = wmma_bf16(a, bm, acc[j]);
        }
    }
    for (int j = 0; j < 4; ++j) {
        int o = obase + j * 16 + col;
        float bias = (o < KD) ? bk[o] : bv[o - KD];
        for (int r = 0; r < 8; ++r) {
            size_t p = (size_t)(p0 + rbase + r);
            float v = acc[j][r] + bias;
            if (o < KD) qk[p * KD + o] = (__bf16)fmaxf(v, 0.f);
            else        valp[p * VD + (o - KD)] = (__bf16)v;
        }
    }
}

// ---------------------------------------------------------------------------
// 3) value [pix][v] -> [v][pix] transpose (per batch), LDS 32x32 tiles
//    grid(NPIX/32, VD/32, 8), block 256
// ---------------------------------------------------------------------------
__global__ void transpose_val(const __bf16* __restrict__ vp, __bf16* __restrict__ vt) {
    __shared__ __bf16 t[32][33];
    int tx = threadIdx.x & 31, ty = threadIdx.x >> 5;   // ty 0..7
    int n0 = blockIdx.x * 32, v0 = blockIdx.y * 32, b = blockIdx.z;
    const __bf16* src = vp + (size_t)b * NPIX * VD;
    for (int i = 0; i < 4; ++i)
        t[ty + i * 8][tx] = src[(size_t)(n0 + ty + i * 8) * VD + v0 + tx];
    __syncthreads();
    __bf16* dst = vt + (size_t)b * VD * NPIX;
    for (int i = 0; i < 4; ++i)
        dst[(size_t)(v0 + ty + i * 8) * NPIX + n0 + tx] = t[tx][ty + i * 8];
}

// ---------------------------------------------------------------------------
// 4) Fused attention: per block 16 query rows; S row-panel in LDS (bf16,
//    padded stride 2312), softmax in LDS, P @ V^T with WMMA; /rowsum folded
//    into epilogue.  grid(NPIX/16, 8), block 128 (4 waves), dyn LDS ~74.6 KB
// ---------------------------------------------------------------------------
#define SPAD 2312
__global__ void attention_kernel(const __bf16* __restrict__ qk,
                                 const __bf16* __restrict__ val,
                                 __bf16* __restrict__ ctx) {
    extern __shared__ char smem[];
    __bf16* P     = (__bf16*)smem;                       // [16][SPAD]
    float* red    = (float*)(smem + (size_t)16 * SPAD * 2);
    float* rowmax = red + 128;
    float* rsum   = rowmax + 16;

    int lane = threadIdx.x & 31, w = threadIdx.x >> 5;
    int b = blockIdx.y, n0 = blockIdx.x * 16;
    int col = lane & 15;
    int h8 = (lane < 16) ? 0 : 8;
    int rbase = (lane >> 4) << 3;
    const __bf16* qbase = qk + (size_t)b * NPIX * KD;

    // Resident query A-fragments (16 rows x K=256)
    Frag aq[8];
    {
        const __bf16* arow = qbase + (size_t)(n0 + col) * KD + h8;
        for (int kt = 0; kt < 8; ++kt) aq[kt] = load_frag(arow + kt * 32);
    }

    // S = (Q K^T) / 16  -> LDS (bf16), waves stripe the m dimension
    for (int mt = w * 16; mt < NPIX; mt += 64) {
        const __bf16* brow = qbase + (size_t)(mt + col) * KD + h8;
        v8f c = {};
        for (int kt = 0; kt < 8; ++kt) {
            Frag bm = load_frag(brow + kt * 32);
            c = wmma_bf16(aq[kt], bm, c);
        }
        for (int r = 0; r < 8; ++r)
            P[(size_t)(rbase + r) * SPAD + mt + col] = (__bf16)(c[r] * 0.0625f);
    }
    __syncthreads();

    // softmax over m: 8 threads per row
    int row = threadIdx.x >> 3, sub = threadIdx.x & 7;
    float lmax = -3.4e38f;
    for (int m = sub; m < NPIX; m += 8)
        lmax = fmaxf(lmax, (float)P[(size_t)row * SPAD + m]);
    red[row * 8 + sub] = lmax;
    __syncthreads();
    if (sub == 0) {
        float mx = red[row * 8];
        for (int i = 1; i < 8; ++i) mx = fmaxf(mx, red[row * 8 + i]);
        rowmax[row] = mx;
    }
    __syncthreads();
    float mx = rowmax[row], s = 0.f;
    for (int m = sub; m < NPIX; m += 8) {
        float e = __expf((float)P[(size_t)row * SPAD + m] - mx);
        P[(size_t)row * SPAD + m] = (__bf16)e;
        s += e;
    }
    red[row * 8 + sub] = s;
    __syncthreads();
    if (sub == 0) {
        float t = 0.f;
        for (int i = 0; i < 8; ++i) t += red[row * 8 + i];
        rsum[row] = 1.0f / t;
    }
    __syncthreads();

    // ctx[n][v] = (P @ V^T) * rsum ; wave owns 64 v-columns
    int vb = w * 64;
    const __bf16* vbase = val + (size_t)b * VD * NPIX;
    v8f acc[4] = {};
    for (int mt = 0; mt < NPIX; mt += 32) {
        Frag pa;
        const __bf16* prow = P + (size_t)col * SPAD + mt + h8;
        pa.h[0] = *(const v8bf*)(prow);
        pa.h[1] = *(const v8bf*)(prow + 16);
        for (int j = 0; j < 4; ++j) {
            Frag bm = load_frag(vbase + (size_t)(vb + j * 16 + col) * NPIX + mt + h8);
            acc[j] = wmma_bf16(pa, bm, acc[j]);
        }
    }
    __bf16* cbase = ctx + ((size_t)b * NPIX + n0) * VD;
    for (int j = 0; j < 4; ++j)
        for (int r = 0; r < 8; ++r) {
            int rw = rbase + r;
            cbase[(size_t)rw * VD + vb + j * 16 + col] = (__bf16)(acc[j][r] * rsum[rw]);
        }
}

// ---------------------------------------------------------------------------
// 5) Ww 1x1 conv: ctx [pix][256] -> ctxw fp32 [pix][512]
//    grid(BN_/16, 2), block 128
// ---------------------------------------------------------------------------
__global__ void ww_gemm(const __bf16* __restrict__ ctx, const __bf16* __restrict__ wwf,
                        const float* __restrict__ bw, float* __restrict__ ctxw) {
    int lane = threadIdx.x & 31, w = threadIdx.x >> 5;
    int p0 = blockIdx.x * 16;
    int obase = (blockIdx.y * 4 + w) * 64;
    int col = lane & 15, h8 = (lane < 16) ? 0 : 8, rbase = (lane >> 4) << 3;
    const __bf16* arow = ctx + (size_t)(p0 + col) * VD + h8;
    v8f acc[4] = {};
    for (int kt = 0; kt < VD / 32; ++kt) {
        Frag a = load_frag(arow + kt * 32);
        for (int j = 0; j < 4; ++j) {
            Frag bm = load_frag(wwf + (size_t)(obase + j * 16 + col) * VD + kt * 32 + h8);
            acc[j] = wmma_bf16(a, bm, acc[j]);
        }
    }
    for (int j = 0; j < 4; ++j) {
        int o = obase + j * 16 + col;
        float bias = bw[o];
        for (int r = 0; r < 8; ++r)
            ctxw[(size_t)(p0 + rbase + r) * OD + o] = acc[j][r] + bias;
    }
}

// ---------------------------------------------------------------------------
// 6) Bilinear x2 upsample (align_corners=True): ctxw [b][48*48][512] fp32
//    -> up [b][96*96][512] bf16.  grid(9216, 8), block 256 (2 channels/thread)
// ---------------------------------------------------------------------------
__global__ void upsample_kernel(const float* __restrict__ ctxw, __bf16* __restrict__ up) {
    int pix = blockIdx.x, b = blockIdx.y;
    int o = threadIdx.x * 2;
    int Y = pix / WIN, X = pix % WIN;
    const float sc = 47.0f / 95.0f;
    float yf = Y * sc, xf = X * sc;
    int y0 = (int)yf, x0 = (int)xf;
    int y1 = min(y0 + 1, HP - 1), x1 = min(x0 + 1, WP - 1);
    float wy = yf - y0, wx = xf - x0;
    const float* base = ctxw + (size_t)b * NPIX * OD;
    float2 c00 = *(const float2*)(base + (size_t)(y0 * WP + x0) * OD + o);
    float2 c01 = *(const float2*)(base + (size_t)(y0 * WP + x1) * OD + o);
    float2 c10 = *(const float2*)(base + (size_t)(y1 * WP + x0) * OD + o);
    float2 c11 = *(const float2*)(base + (size_t)(y1 * WP + x1) * OD + o);
    float t0 = c00.x + (c01.x - c00.x) * wx;
    float b0 = c10.x + (c11.x - c10.x) * wx;
    float t1 = c00.y + (c01.y - c00.y) * wx;
    float b1 = c10.y + (c11.y - c10.y) * wx;
    __bf16* dst = up + ((size_t)b * HWOUT + pix) * OD + o;
    dst[0] = (__bf16)(t0 + (b0 - t0) * wy);
    dst[1] = (__bf16)(t1 + (b1 - t1) * wy);
}

// ---------------------------------------------------------------------------
// 7) Final 512x512 conv + folded BN + ReLU -> NCHW fp32 out.
//    Wave = 16 pixels x 128 out-channels; epilogue transposed through LDS so
//    NCHW stores are 64B-contiguous.  grid(BHW/64, 4), block 128
// ---------------------------------------------------------------------------
__global__ void final_gemm(const __bf16* __restrict__ up, const __bf16* __restrict__ wrf,
                           const float* __restrict__ br, float* __restrict__ out) {
    __shared__ float tile[4][16][17];
    int lane = threadIdx.x & 31, w = threadIdx.x >> 5;
    int p0 = blockIdx.x * 64 + w * 16;
    int obase = blockIdx.y * 128;
    int col = lane & 15, h8 = (lane < 16) ? 0 : 8, rbase = (lane >> 4) << 3;
    const __bf16* arow = up + (size_t)(p0 + col) * OD + h8;
    v8f acc[8] = {};
    for (int kt = 0; kt < OD / 32; ++kt) {
        Frag a = load_frag(arow + kt * 32);
        if (kt + 1 < OD / 32) __builtin_prefetch(arow + (kt + 1) * 32, 0, 0);
        for (int j = 0; j < 8; ++j) {
            Frag bm = load_frag(wrf + (size_t)(obase + j * 16 + col) * OD + kt * 32 + h8);
            acc[j] = wmma_bf16(a, bm, acc[j]);
        }
    }
    int b = p0 / HWOUT, pb = p0 % HWOUT;
    for (int j = 0; j < 8; ++j) {
        float bias = br[obase + j * 16 + col];
        for (int r = 0; r < 8; ++r)
            tile[w][col][rbase + r] = fmaxf(acc[j][r] + bias, 0.f);  // [o_local][pix]
        asm volatile("s_wait_dscnt 0" ::: "memory");
        for (int rr = 0; rr < 8; ++rr) {
            int ol = rr * 2 + (lane >> 4);
            float v = tile[w][ol][col];
            out[((size_t)b * OD + obase + j * 16 + ol) * HWOUT + pb + col] = v;
        }
        asm volatile("s_wait_dscnt 0" ::: "memory");
    }
}

// ---------------------------------------------------------------------------
// Launch
// ---------------------------------------------------------------------------
extern "C" void kernel_launch(void* const* d_in, const int* in_sizes, int n_in,
                              void* d_out, int out_size, void* d_ws, size_t ws_size,
                              hipStream_t stream) {
    (void)in_sizes; (void)n_in; (void)out_size; (void)ws_size;
    const float* x  = (const float*)d_in[0];
    const float* Wk = (const float*)d_in[1];
    const float* kg = (const float*)d_in[2];
    const float* kb = (const float*)d_in[3];
    const float* km = (const float*)d_in[4];
    const float* kv = (const float*)d_in[5];
    const float* Wv = (const float*)d_in[6];
    const float* bv = (const float*)d_in[7];
    const float* Ww = (const float*)d_in[8];
    const float* bw = (const float*)d_in[9];
    const float* Wr = (const float*)d_in[10];
    const float* rg = (const float*)d_in[11];
    const float* rb = (const float*)d_in[12];
    const float* rm = (const float*)d_in[13];
    const float* rv = (const float*)d_in[14];
    float* out = (float*)d_out;

    char* ws = (char*)d_ws;
    __bf16* xp   = (__bf16*)(ws + OFF_XP);
    __bf16* qk   = (__bf16*)(ws + OFF_QK);
    __bf16* valp = (__bf16*)(ws + OFF_VALP);
    __bf16* valt = (__bf16*)(ws + OFF_VALT);
    __bf16* ctx  = (__bf16*)(ws + OFF_CTX);
    float*  ctxw = (float*) (ws + OFF_CTXW);
    __bf16* upb  = (__bf16*)(ws + OFF_UP);
    __bf16* wkf  = (__bf16*)(ws + OFF_WKF);
    __bf16* wvf  = (__bf16*)(ws + OFF_WVF);
    __bf16* wwf  = (__bf16*)(ws + OFF_WWF);
    __bf16* wrf  = (__bf16*)(ws + OFF_WRF);
    float*  bk   = (float*) (ws + OFF_BK);
    float*  br   = (float*) (ws + OFF_BR);

    prep_weights<<<1024, 256, 0, stream>>>(Wk, kg, kb, km, kv, Wv, Ww, Wr, rg, rb, rm, rv,
                                           wkf, wvf, wwf, wrf, bk, br);
    pool_kernel<<<dim3(3, 48, 256), 256, 0, stream>>>(x, xp);
    qkv_gemm<<<dim3(BN_ / 16, 2), 128, 0, stream>>>(xp, wkf, wvf, bk, bv, qk, valp);
    transpose_val<<<dim3(NPIX / 32, VD / 32, BATCH), 256, 0, stream>>>(valp, valt);
    size_t attn_lds = (size_t)16 * SPAD * 2 + 128 * 4 + 16 * 4 + 16 * 4;
    attention_kernel<<<dim3(NPIX / 16, BATCH), 128, attn_lds, stream>>>(qk, valt, ctx);
    ww_gemm<<<dim3(BN_ / 16, 2), 128, 0, stream>>>(ctx, wwf, bw, ctxw);
    upsample_kernel<<<dim3(HWOUT, BATCH), 256, 0, stream>>>(ctxw, upb);
    final_gemm<<<dim3(BHW / 64, 4), 128, 0, stream>>>(upb, wrf, br, out);
}
